// SparseSubdivideBlock3d_23828478558287
// MI455X (gfx1250) — matrix operand order repaired and checked by reading, not compiled, not run on hardware
//
#include <hip/hip_runtime.h>
#include <hip/hip_bf16.h>

// ---- types for WMMA fragments -------------------------------------------
typedef __attribute__((ext_vector_type(16))) _Float16 v16h;
typedef __attribute__((ext_vector_type(8)))  _Float16 v8h;
typedef __attribute__((ext_vector_type(8)))  float    v8f;
typedef __fp16 fp16x2 __attribute__((ext_vector_type(2)));

union V16H { v16h v; _Float16 h[16]; fp16x2 p[8]; v8h q[2]; };

#define NVOX   20000
#define NVOX2  (8 * NVOX)
#define CIN    64
#define COUT   64
#define D1     96
#define D2     192
#define TILES1 (NVOX / 16)     // 1250
#define TILES2 (NVOX2 / 16)    // 10000
#define UNITS1 ((TILES1 + 1) / 2)   // 625  (2 tiles per wave)
#define UNITS2 (TILES2 / 2)         // 5000
#define KCHUNK 14                   // k-offsets staged in LDS at a time
#define HALVES_PER_K  4096          // 8 frags * 512 halves
#define SB_HALVES (KCHUNK * HALVES_PER_K)   // 57344 halves = 112 KB

// 27-neighborhood offsets, product([-1,0,1]^3) order
__device__ __constant__ int OFFD0[27] = {-1,-1,-1,-1,-1,-1,-1,-1,-1, 0,0,0,0,0,0,0,0,0, 1,1,1,1,1,1,1,1,1};
__device__ __constant__ int OFFD1[27] = {-1,-1,-1,0,0,0,1,1,1, -1,-1,-1,0,0,0,1,1,1, -1,-1,-1,0,0,0,1,1,1};
__device__ __constant__ int OFFD2[27] = {-1,0,1,-1,0,1,-1,0,1, -1,0,1,-1,0,1,-1,0,1, -1,0,1,-1,0,1,-1,0,1};

// ---- grid init -----------------------------------------------------------
__global__ void k_init_grid(int* __restrict__ g, int n) {
  int i = blockIdx.x * blockDim.x + threadIdx.x;
  if (i < n) g[i] = -1;
}

// ---- scatter voxel index into dense lookup grids -------------------------
__global__ void k_scatter1(const int* __restrict__ coords, int* __restrict__ grid) {
  int i = blockIdx.x * blockDim.x + threadIdx.x;
  if (i >= NVOX) return;
  int c0 = coords[3 * i], c1 = coords[3 * i + 1], c2 = coords[3 * i + 2];
  grid[(c0 * D1 + c1) * D1 + c2] = i;
}

__global__ void k_scatter2(const int* __restrict__ coords, int* __restrict__ grid) {
  int j = blockIdx.x * blockDim.x + threadIdx.x;
  if (j >= NVOX2) return;
  int p = j >> 3, ch = j & 7;
  int c0 = coords[3 * p]     * 2 + ((ch >> 2) & 1);
  int c1 = coords[3 * p + 1] * 2 + ((ch >> 1) & 1);
  int c2 = coords[3 * p + 2] * 2 + (ch & 1);
  grid[(c0 * D2 + c1) * D2 + c2] = j;
}

// ---- one-shot f32 -> f16 conversion of the input feature matrix ----------
__global__ void k_prep_f(const float* __restrict__ feats, _Float16* __restrict__ out) {
  int i = blockIdx.x * blockDim.x + threadIdx.x;
  if (i >= NVOX * CIN / 4) return;
  float4 f = *(const float4*)(feats + (size_t)i * 4);
  fp16x2* o = (fp16x2*)(out + (size_t)i * 4);
  o[0] = __builtin_amdgcn_cvt_pkrtz(f.x, f.y);
  o[1] = __builtin_amdgcn_cvt_pkrtz(f.z, f.w);
}

// ---- convert f32 weights into f16 WMMA B-fragment layout -----------------
// Fragment (k27, kc, nt): lane L holds column nt*16+(L&15); half h -> K = kc*32 + (L>>4)*16 + h
__global__ void k_prep_w(const float* __restrict__ W1, const float* __restrict__ W2,
                         _Float16* __restrict__ Wb1, _Float16* __restrict__ Wb2) {
  int t = blockIdx.x * blockDim.x + threadIdx.x;
  const int per = 27 * 8 * 32;
  if (t >= 2 * per) return;
  const float* W = W1;
  _Float16* Wb = Wb1;
  int u = t;
  if (u >= per) { u -= per; W = W2; Wb = Wb2; }
  int lane = u & 31;
  int frag = u >> 5;                 // ((k27*2+kc)*4+nt)
  int k27 = frag >> 3;
  int kc  = (frag >> 2) & 1;
  int nt  = frag & 3;
  int col = nt * 16 + (lane & 15);
  int g   = lane >> 4;
  _Float16* dst = Wb + (frag * 32 + lane) * 16;
#pragma unroll
  for (int h = 0; h < 16; ++h) {
    int kin = kc * 32 + g * 16 + h;
    dst[h] = (_Float16)W[(k27 * 64 + kin) * 64 + col];
  }
}

__device__ __forceinline__ float silu(float x) {
  return x / (1.0f + __expf(-x));
}

// gather a 16x32 f16 A-fragment slice; feature row = (idx >> shift)
__device__ __forceinline__ void gatherA_f16(V16H& A, const _Float16* __restrict__ hfeat,
                                            int idx, int shift, int kc, int g) {
  if (idx >= 0) {
    const _Float16* fp = hfeat + (size_t)(idx >> shift) * COUT + kc * 32 + g * 16;
    A.q[0] = *(const v8h*)(fp);        // K = kc*32 + g*16 + 0..7
    A.q[1] = *(const v8h*)(fp + 8);    // K = kc*32 + g*16 + 8..15
  } else {
#pragma unroll
    for (int i = 0; i < 16; ++i) A.h[i] = (_Float16)0.0f;
  }
}

// ---- layer 1: gather-GEMM submanifold conv (f16 feats -> f16 h) ----------
// One wave computes a 32-row x 64-col output tile (2 M-tiles share each B frag).
__global__ __launch_bounds__(256)
void k_conv1(const int* __restrict__ coords, const _Float16* __restrict__ feats16,
             const int* __restrict__ grid, const _Float16* __restrict__ Wb,
             const float* __restrict__ bias, _Float16* __restrict__ out) {
  __shared__ __align__(16) _Float16 sB[SB_HALVES];
  const int t    = threadIdx.x;
  const int lane = t & 31;
  const int wave = t >> 5;
  int unit = blockIdx.x * 8 + wave;
  if (unit >= UNITS1) unit = UNITS1 - 1;    // clamp (duplicates rewrite same data)
  const int tile0 = unit * 2;
  const int rcol = lane & 15;
  const int g    = lane >> 4;
  const int r0 = tile0 * 16 + rcol;
  const int r1 = r0 + 16;

  const int a0 = coords[3 * r0], a1 = coords[3 * r0 + 1], a2 = coords[3 * r0 + 2];
  const int b0 = coords[3 * r1], b1c = coords[3 * r1 + 1], b2 = coords[3 * r1 + 2];

  v8f acc[8];   // [tile 0..1][nt 0..3]
#pragma unroll
  for (int nt = 0; nt < 4; ++nt) {
    float bv = bias[nt * 16 + rcol];
#pragma unroll
    for (int v = 0; v < 8; ++v) { acc[nt][v] = bv; acc[4 + nt][v] = bv; }
  }

  for (int kbase = 0; kbase < 27; kbase += KCHUNK) {
    const int len = (27 - kbase) < KCHUNK ? (27 - kbase) : KCHUNK;
    __syncthreads();
    {   // stage this chunk's B fragments into LDS
      const float4* src = (const float4*)(Wb + kbase * HALVES_PER_K);
      float4* dst = (float4*)sB;
      const int n4 = len * (HALVES_PER_K / 8);   // float4 count
      for (int i = t; i < n4; i += 256) dst[i] = src[i];
    }
    __syncthreads();

    for (int kk = 0; kk < len; ++kk) {
      const int k = kbase + kk;
      const int d0 = OFFD0[k], d1 = OFFD1[k], d2 = OFFD2[k];

      int nA0 = a0 + d0, nA1 = a1 + d1, nA2 = a2 + d2;
      int nB0 = b0 + d0, nB1 = b1c + d1, nB2 = b2 + d2;
      int idxA = -1, idxB = -1;
      if (((unsigned)nA0 < D1) & ((unsigned)nA1 < D1) & ((unsigned)nA2 < D1))
        idxA = grid[(nA0 * D1 + nA1) * D1 + nA2];
      if (((unsigned)nB0 < D1) & ((unsigned)nB1 < D1) & ((unsigned)nB2 < D1))
        idxB = grid[(nB0 * D1 + nB1) * D1 + nB2];

#pragma unroll
      for (int kc = 0; kc < 2; ++kc) {
        V16H A0, A1;
        gatherA_f16(A0, feats16, idxA, 0, kc, g);
        gatherA_f16(A1, feats16, idxB, 0, kc, g);
        const _Float16* lb = sB + ((kk * 2 + kc) * 4) * 512 + lane * 16;
#pragma unroll
        for (int nt = 0; nt < 4; ++nt) {
          V16H B;
          B.v = *(const v16h*)(lb + nt * 512);
          acc[nt] = __builtin_amdgcn_wmma_f32_16x16x32_f16(
              false, A0.v, false, B.v, (short)0, acc[nt], false, false);
          acc[4 + nt] = __builtin_amdgcn_wmma_f32_16x16x32_f16(
              false, A1.v, false, B.v, (short)0, acc[4 + nt], false, false);
        }
      }
    }
  }

#pragma unroll
  for (int tt = 0; tt < 2; ++tt) {
#pragma unroll
    for (int nt = 0; nt < 4; ++nt) {
#pragma unroll
      for (int v = 0; v < 8; ++v) {
        int r = (tile0 + tt) * 16 + v + g * 8;
        out[r * COUT + nt * 16 + rcol] = (_Float16)silu(acc[tt * 4 + nt][v]);
      }
    }
  }
}

// ---- layer 2: conv at 2x resolution (f16 h -> f32 out) -------------------
__global__ __launch_bounds__(256)
void k_conv2(const int* __restrict__ coords, const _Float16* __restrict__ hfeat,
             const int* __restrict__ grid, const _Float16* __restrict__ Wb,
             const float* __restrict__ bias, float* __restrict__ out) {
  __shared__ __align__(16) _Float16 sB[SB_HALVES];
  const int t    = threadIdx.x;
  const int lane = t & 31;
  const int wave = t >> 5;
  int unit = blockIdx.x * 8 + wave;
  if (unit >= UNITS2) unit = UNITS2 - 1;
  const int tile0 = unit * 2;
  const int rcol = lane & 15;
  const int g    = lane >> 4;
  const int r0 = tile0 * 16 + rcol;   // fine voxel j for tile 0
  const int r1 = r0 + 16;

  int pA = r0 >> 3, cA = r0 & 7;
  int pB = r1 >> 3, cB = r1 & 7;
  const int a0 = coords[3 * pA]     * 2 + ((cA >> 2) & 1);
  const int a1 = coords[3 * pA + 1] * 2 + ((cA >> 1) & 1);
  const int a2 = coords[3 * pA + 2] * 2 + (cA & 1);
  const int b0 = coords[3 * pB]     * 2 + ((cB >> 2) & 1);
  const int b1c = coords[3 * pB + 1] * 2 + ((cB >> 1) & 1);
  const int b2 = coords[3 * pB + 2] * 2 + (cB & 1);

  v8f acc[8];
#pragma unroll
  for (int nt = 0; nt < 4; ++nt) {
    float bv = bias[nt * 16 + rcol];
#pragma unroll
    for (int v = 0; v < 8; ++v) { acc[nt][v] = bv; acc[4 + nt][v] = bv; }
  }

  for (int kbase = 0; kbase < 27; kbase += KCHUNK) {
    const int len = (27 - kbase) < KCHUNK ? (27 - kbase) : KCHUNK;
    __syncthreads();
    {
      const float4* src = (const float4*)(Wb + kbase * HALVES_PER_K);
      float4* dst = (float4*)sB;
      const int n4 = len * (HALVES_PER_K / 8);
      for (int i = t; i < n4; i += 256) dst[i] = src[i];
    }
    __syncthreads();

    for (int kk = 0; kk < len; ++kk) {
      const int k = kbase + kk;
      const int d0 = OFFD0[k], d1 = OFFD1[k], d2 = OFFD2[k];

      int nA0 = a0 + d0, nA1 = a1 + d1, nA2 = a2 + d2;
      int nB0 = b0 + d0, nB1 = b1c + d1, nB2 = b2 + d2;
      int idxA = -1, idxB = -1;
      if (((unsigned)nA0 < D2) & ((unsigned)nA1 < D2) & ((unsigned)nA2 < D2))
        idxA = grid[(nA0 * D2 + nA1) * D2 + nA2];
      if (((unsigned)nB0 < D2) & ((unsigned)nB1 < D2) & ((unsigned)nB2 < D2))
        idxB = grid[(nB0 * D2 + nB1) * D2 + nB2];

#pragma unroll
      for (int kc = 0; kc < 2; ++kc) {
        V16H A0, A1;
        gatherA_f16(A0, hfeat, idxA, 3, kc, g);
        gatherA_f16(A1, hfeat, idxB, 3, kc, g);
        const _Float16* lb = sB + ((kk * 2 + kc) * 4) * 512 + lane * 16;
#pragma unroll
        for (int nt = 0; nt < 4; ++nt) {
          V16H B;
          B.v = *(const v16h*)(lb + nt * 512);
          acc[nt] = __builtin_amdgcn_wmma_f32_16x16x32_f16(
              false, A0.v, false, B.v, (short)0, acc[nt], false, false);
          acc[4 + nt] = __builtin_amdgcn_wmma_f32_16x16x32_f16(
              false, A1.v, false, B.v, (short)0, acc[4 + nt], false, false);
        }
      }
    }
  }

#pragma unroll
  for (int tt = 0; tt < 2; ++tt) {
#pragma unroll
    for (int nt = 0; nt < 4; ++nt) {
#pragma unroll
      for (int v = 0; v < 8; ++v) {
        int r = (tile0 + tt) * 16 + v + g * 8;
        out[r * COUT + nt * 16 + rcol] = silu(acc[tt * 4 + nt][v]);
      }
    }
  }
}

// ---- host-side launcher --------------------------------------------------
extern "C" void kernel_launch(void* const* d_in, const int* in_sizes, int n_in,
                              void* d_out, int out_size, void* d_ws, size_t ws_size,
                              hipStream_t stream) {
  (void)in_sizes; (void)n_in; (void)out_size; (void)ws_size;
  const int*   coords = (const int*)d_in[0];
  const float* feats  = (const float*)d_in[1];
  const float* W1     = (const float*)d_in[2];
  const float* b1     = (const float*)d_in[3];
  const float* W2     = (const float*)d_in[4];
  const float* b2     = (const float*)d_in[5];

  char* ws = (char*)d_ws;
  const size_t GRID1_B = (size_t)D1 * D1 * D1 * 4;        // 3,538,944
  const size_t GRID2_B = (size_t)D2 * D2 * D2 * 4;        // 28,311,552
  const size_t HF_B    = (size_t)NVOX * COUT * 2;         // 2,560,000
  const size_t WB_B    = (size_t)27 * 8 * 32 * 16 * 2;    // 221,184
  const size_t F16_B   = (size_t)NVOX * CIN * 2;          // 2,560,000

  int*      grid1   = (int*)(ws);
  int*      grid2   = (int*)(ws + GRID1_B);
  _Float16* hfeat   = (_Float16*)(ws + GRID1_B + GRID2_B);
  _Float16* Wb1     = (_Float16*)(ws + GRID1_B + GRID2_B + HF_B);
  _Float16* Wb2     = (_Float16*)(ws + GRID1_B + GRID2_B + HF_B + WB_B);
  _Float16* feats16 = (_Float16*)(ws + GRID1_B + GRID2_B + HF_B + 2 * WB_B);
  (void)F16_B;

  const int n1 = D1 * D1 * D1;
  const int n2 = D2 * D2 * D2;
  k_init_grid<<<(n1 + 255) / 256, 256, 0, stream>>>(grid1, n1);
  k_init_grid<<<(n2 + 255) / 256, 256, 0, stream>>>(grid2, n2);
  k_scatter1<<<(NVOX + 255) / 256, 256, 0, stream>>>(coords, grid1);
  k_scatter2<<<(NVOX2 + 255) / 256, 256, 0, stream>>>(coords, grid2);
  k_prep_f<<<(NVOX * CIN / 4 + 255) / 256, 256, 0, stream>>>(feats, feats16);
  k_prep_w<<<(2 * 27 * 8 * 32 + 255) / 256, 256, 0, stream>>>(W1, W2, Wb1, Wb2);
  k_conv1<<<(UNITS1 + 7) / 8, 256, 0, stream>>>(coords, feats16, grid1, Wb1, b1, hfeat);
  k_conv2<<<UNITS2 / 8, 256, 0, stream>>>(coords, hfeat, grid2, Wb2, b2, (float*)d_out);
}
